// DecoderRNN_4131758539049
// MI455X (gfx1250) — compile-verified, matrix operand order checked by reading, not run on hardware
//
#include <hip/hip_runtime.h>
#include <hip/hip_bf16.h>

typedef __bf16 bf16_t;
typedef __attribute__((ext_vector_type(16))) __bf16 v16bf;
typedef __attribute__((ext_vector_type(8)))  __bf16 v8bf;
typedef __attribute__((ext_vector_type(8)))  float  v8f;

#define B_   512
#define S_   200
#define V_   100

// ---------------- helper kernels ----------------

__global__ void k_cvt_bf16(const float* __restrict__ in, bf16_t* __restrict__ out, int n) {
  int i = blockIdx.x * blockDim.x + threadIdx.x;
  if (i < n) out[i] = (bf16_t)in[i];
}

// out_w [100,2048] -> bf16 padded [128,2048] (zero fill)
__global__ void k_pad_outw(const float* __restrict__ w, bf16_t* __restrict__ wb) {
  int idx = blockIdx.x * blockDim.x + threadIdx.x;   // 128*2048
  int n = idx >> 11;
  int k = idx & 2047;
  float v = (n < V_) ? w[(size_t)n * 2048 + k] : 0.0f;
  wb[idx] = (bf16_t)v;
}

__global__ void k_pad_outb(const float* __restrict__ b, float* __restrict__ bp) {
  int i = threadIdx.x; // 128 threads
  bp[i] = (i < V_) ? b[i] : 0.0f;
}

// decode_embed = concat(z, cond); also write the constant tails of x0 and xout
__global__ void k_build_de(const float* __restrict__ z, const float* __restrict__ c,
                           bf16_t* __restrict__ de, bf16_t* __restrict__ x0,
                           bf16_t* __restrict__ xout) {
  int idx = blockIdx.x * blockDim.x + threadIdx.x;  // 512*1024
  int b = idx >> 10, j = idx & 1023;
  float v = (j < 512) ? z[(size_t)b * 512 + j] : c[(size_t)b * 512 + (j - 512)];
  bf16_t bv = (bf16_t)v;
  de[idx] = bv;
  x0[(size_t)b * 1536 + 512 + j]   = bv;
  xout[(size_t)b * 2048 + 1024 + j] = bv;
}

// replicate h0 (f32) to layers 1,2 and make bf16 copies (layer2 bf16 lives in xout[:, :1024])
__global__ void k_rep_h(const float* __restrict__ h0,
                        float* __restrict__ h1f, float* __restrict__ h2f,
                        bf16_t* __restrict__ hb0, bf16_t* __restrict__ hb1,
                        bf16_t* __restrict__ xout) {
  int idx = blockIdx.x * blockDim.x + threadIdx.x; // 512*1024
  float v = h0[idx];
  h1f[idx] = v; h2f[idx] = v;
  bf16_t bv = (bf16_t)v;
  hb0[idx] = bv; hb1[idx] = bv;
  int b = idx >> 10, j = idx & 1023;
  xout[(size_t)b * 2048 + j] = bv;
}

// teacher-forced token embedding gather into x0[:, :512], 8 halves per thread
__global__ void k_embed(const int* __restrict__ toks, int t,
                        const bf16_t* __restrict__ embb, bf16_t* __restrict__ x0) {
  int idx = blockIdx.x * blockDim.x + threadIdx.x; // 512*64
  int b = idx >> 6, j = (idx & 63) << 3;
  int tok = (t == 0) ? 1 : toks[(size_t)b * S_ + (t - 1)];
  *(v8bf*)&x0[(size_t)b * 1536 + j] = *(const v8bf*)&embb[(size_t)tok * 512 + j];
}

// fused GRU gate: h = (1-z)*tanh(i_n + r*h_n) + z*h_prev  (in-place f32 + bf16 copy)
__global__ void k_gru_gate(const float* __restrict__ gi, const float* __restrict__ gh,
                           const float* __restrict__ hprev, float* __restrict__ hnew,
                           bf16_t* __restrict__ hb, int hb_stride) {
  int idx = blockIdx.x * blockDim.x + threadIdx.x; // 512*1024
  int b = idx >> 10, j = idx & 1023;
  size_t g = (size_t)b * 3072 + j;
  float r  = 1.0f / (1.0f + __expf(-(gi[g]        + gh[g]       )));
  float zg = 1.0f / (1.0f + __expf(-(gi[g + 1024] + gh[g + 1024])));
  float n  = tanhf(gi[g + 2048] + r * gh[g + 2048]);
  float h  = (1.0f - zg) * n + zg * hprev[idx];
  hnew[idx] = h;
  hb[(size_t)b * hb_stride + j] = (bf16_t)h;
}

// ---------------- WMMA GEMM: C[M,N] = A[M,K](bf16) * W[N,K]^T(bf16) + bias ----------------
// block = 256 threads = 8 wave32; block tile 128x128; wave tile 32x64 (2x4 frags).
// K panel = 64, double-buffered LDS filled with GLOBAL_LOAD_ASYNC_TO_LDS_B128 (ASYNCcnt).

#define TM 128
#define TN 128
#define TK 64
#define LSTR 72   // 64 + 8 halves pad: 144B row stride, 16B aligned, conflict-free ds_load_b128

__global__ void __launch_bounds__(256) k_gemm_bf16(
    const bf16_t* __restrict__ A, int lda,
    const bf16_t* __restrict__ W, int ldw,
    const float* __restrict__ bias,
    float* __restrict__ C, long long ldc, int nvalid, int K) {
  __shared__ __align__(16) bf16_t lA[2][TM][LSTR];
  __shared__ __align__(16) bf16_t lB[2][TN][LSTR];

  const int tid  = threadIdx.x;
  const int lane = tid & 31;
  const int w    = tid >> 5;          // wave 0..7
  const int wm   = w >> 1;            // 0..3 : 32-row slab
  const int wn   = w & 1;             // 0..1 : 64-col slab
  const int m0   = blockIdx.y * TM;
  const int n0   = blockIdx.x * TN;

  const int lrow = lane & 15;
  const int kh   = (lane >> 4) << 3;  // halves offset 0 or 8 (ISA 16-bit frag layout)

  v8f acc[2][4] = {};

  // stage one 128x64 A panel + 128x64 W panel into LDS buffer `buf`
  // 1024 chunks of 16B per tile; 256 threads -> 4 A + 4 B chunks each
  // -> 8 async instructions per wave per stage (ASYNCcnt += 8)
  auto stage = [&](int buf, int k0) {
#pragma unroll
    for (int i = 0; i < 4; ++i) {
      int c   = tid + i * 256;        // 0..1023
      int row = c >> 3;
      int ch  = (c & 7) << 3;         // halves within panel
      unsigned la = (unsigned)(unsigned long long)(const void*)&lA[buf][row][ch];
      unsigned long long ga =
          (unsigned long long)(const void*)(A + (size_t)(m0 + row) * lda + k0 + ch);
      asm volatile("global_load_async_to_lds_b128 %0, %1, off"
                   :: "v"(la), "v"(ga) : "memory");
      unsigned lb = (unsigned)(unsigned long long)(const void*)&lB[buf][row][ch];
      unsigned long long gb =
          (unsigned long long)(const void*)(W + (size_t)(n0 + row) * ldw + k0 + ch);
      asm volatile("global_load_async_to_lds_b128 %0, %1, off"
                   :: "v"(lb), "v"(gb) : "memory");
    }
  };

  stage(0, 0);
  const int nstages = K / TK;
  for (int s = 0; s < nstages; ++s) {
    const int buf = s & 1;
    if (s + 1 < nstages) {
      stage(buf ^ 1, (s + 1) * TK);
      asm volatile("s_wait_asynccnt 0x8" ::: "memory");  // current panel complete (in-order)
    } else {
      asm volatile("s_wait_asynccnt 0x0" ::: "memory");
    }
    __syncthreads();   // current panel visible to all waves

#pragma unroll
    for (int ks = 0; ks < 2; ++ks) {
      const int ko = ks * 32 + kh;
      v16bf afr[2], bfr[4];
#pragma unroll
      for (int i = 0; i < 2; ++i) {
        int r = wm * 32 + i * 16 + lrow;
        v8bf lo = *(const v8bf*)&lA[buf][r][ko];
        v8bf hi = *(const v8bf*)&lA[buf][r][ko + 16];
#pragma unroll
        for (int e = 0; e < 8; ++e) { afr[i][e] = lo[e]; afr[i][8 + e] = hi[e]; }
      }
#pragma unroll
      for (int j = 0; j < 4; ++j) {
        int r = wn * 64 + j * 16 + lrow;
        v8bf lo = *(const v8bf*)&lB[buf][r][ko];
        v8bf hi = *(const v8bf*)&lB[buf][r][ko + 16];
#pragma unroll
        for (int e = 0; e < 8; ++e) { bfr[j][e] = lo[e]; bfr[j][8 + e] = hi[e]; }
      }
#pragma unroll
      for (int i = 0; i < 2; ++i)
#pragma unroll
        for (int j = 0; j < 4; ++j)
          acc[i][j] = __builtin_amdgcn_wmma_f32_16x16x32_bf16(
              false, afr[i], false, bfr[j], (short)0, acc[i][j], false, false);
    }
    __syncthreads();   // all waves done reading `buf` before it is refilled
  }

  // epilogue: C/D layout -> lane<16: rows m0+r, col=lane ; lane>=16: rows m0+8+r
  const int colb = n0 + wn * 64 + (lane & 15);
  const int rb   = m0 + wm * 32 + ((lane >> 4) << 3);
#pragma unroll
  for (int j = 0; j < 4; ++j) {
    int col = colb + j * 16;
    if (col >= nvalid) continue;
    float bv = bias ? bias[col] : 0.0f;
#pragma unroll
    for (int i = 0; i < 2; ++i)
#pragma unroll
      for (int r = 0; r < 8; ++r)
        C[(size_t)(rb + i * 16 + r) * (size_t)ldc + col] = acc[i][j][r] + bv;
  }
}

// ---------------- orchestration ----------------

extern "C" void kernel_launch(void* const* d_in, const int* in_sizes, int n_in,
                              void* d_out, int out_size, void* d_ws, size_t ws_size,
                              hipStream_t stream) {
  (void)in_sizes; (void)n_in; (void)out_size; (void)ws_size;

  const int*   inputs = (const int*)  d_in[0];
  const float* z      = (const float*)d_in[1];
  const float* cond   = (const float*)d_in[2];
  // d_in[3] temperature: >=1 -> teacher forcing always (ignored)
  const float* emb    = (const float*)d_in[4];
  const float* i2h_w  = (const float*)d_in[5];
  const float* i2h_b  = (const float*)d_in[6];
  const float* out_w  = (const float*)d_in[7];
  const float* out_b  = (const float*)d_in[8];
  const float *wih[3], *whh[3], *bih[3], *bhh[3];
  for (int l = 0; l < 3; ++l) {
    wih[l] = (const float*)d_in[9  + 4 * l];
    whh[l] = (const float*)d_in[10 + 4 * l];
    bih[l] = (const float*)d_in[11 + 4 * l];
    bhh[l] = (const float*)d_in[12 + 4 * l];
  }
  float* out = (float*)d_out;

  // bump allocator over workspace (~70 MB)
  char* p = (char*)d_ws;
  auto alloc = [&](size_t bytes) -> void* {
    void* r = (void*)p;
    p += (bytes + 255) & ~(size_t)255;
    return r;
  };

  bf16_t* wihb[3]; bf16_t* whhb[3];
  wihb[0] = (bf16_t*)alloc((size_t)3072 * 1536 * 2);
  whhb[0] = (bf16_t*)alloc((size_t)3072 * 1024 * 2);
  for (int l = 1; l < 3; ++l) {
    wihb[l] = (bf16_t*)alloc((size_t)3072 * 1024 * 2);
    whhb[l] = (bf16_t*)alloc((size_t)3072 * 1024 * 2);
  }
  bf16_t* i2hb  = (bf16_t*)alloc((size_t)1024 * 1024 * 2);
  bf16_t* outwb = (bf16_t*)alloc((size_t)128 * 2048 * 2);
  float*  outbp = (float*) alloc(128 * 4);
  bf16_t* embb  = (bf16_t*)alloc((size_t)V_ * 512 * 2);
  bf16_t* de    = (bf16_t*)alloc((size_t)512 * 1024 * 2);
  bf16_t* x0    = (bf16_t*)alloc((size_t)512 * 1536 * 2);   // [emb | decode_embed]
  bf16_t* xout  = (bf16_t*)alloc((size_t)512 * 2048 * 2);   // [h2  | decode_embed]
  bf16_t* hb0   = (bf16_t*)alloc((size_t)512 * 1024 * 2);
  bf16_t* hb1   = (bf16_t*)alloc((size_t)512 * 1024 * 2);
  float*  hf0   = (float*) alloc((size_t)512 * 1024 * 4);
  float*  hf1   = (float*) alloc((size_t)512 * 1024 * 4);
  float*  hf2   = (float*) alloc((size_t)512 * 1024 * 4);
  float*  gi    = (float*) alloc((size_t)512 * 3072 * 4);
  float*  gh    = (float*) alloc((size_t)512 * 3072 * 4);

  auto cvt = [&](const float* src, bf16_t* dst, size_t n) {
    k_cvt_bf16<<<dim3((unsigned)((n + 255) / 256)), dim3(256), 0, stream>>>(src, dst, (int)n);
  };
  cvt(wih[0], wihb[0], (size_t)3072 * 1536);
  cvt(whh[0], whhb[0], (size_t)3072 * 1024);
  cvt(wih[1], wihb[1], (size_t)3072 * 1024);
  cvt(whh[1], whhb[1], (size_t)3072 * 1024);
  cvt(wih[2], wihb[2], (size_t)3072 * 1024);
  cvt(whh[2], whhb[2], (size_t)3072 * 1024);
  cvt(i2h_w, i2hb, (size_t)1024 * 1024);
  cvt(emb, embb, (size_t)V_ * 512);
  k_pad_outw<<<dim3(1024), dim3(256), 0, stream>>>(out_w, outwb);
  k_pad_outb<<<dim3(1),    dim3(128), 0, stream>>>(out_b, outbp);
  k_build_de<<<dim3(2048), dim3(256), 0, stream>>>(z, cond, de, x0, xout);

  auto gemm = [&](const bf16_t* A, int lda, const bf16_t* W, int ldw,
                  const float* bias, float* C, long long ldc, int nvalid,
                  int N, int K) {
    k_gemm_bf16<<<dim3(N / 128, B_ / 128), dim3(256), 0, stream>>>(
        A, lda, W, ldw, bias, C, ldc, nvalid, K);
  };

  // h0 = decode_embed @ i2h_w^T + i2h_b ; replicate to 3 layers
  gemm(de, 1024, i2hb, 1024, i2h_b, hf0, 1024, 1024, 1024, 1024);
  k_rep_h<<<dim3(2048), dim3(256), 0, stream>>>(hf0, hf1, hf2, hb0, hb1, xout);

  for (int t = 0; t < S_; ++t) {
    k_embed<<<dim3(128), dim3(256), 0, stream>>>(inputs, t, embb, x0);
    // layer 0
    gemm(x0,  1536, wihb[0], 1536, bih[0], gi, 3072, 3072, 3072, 1536);
    gemm(hb0, 1024, whhb[0], 1024, bhh[0], gh, 3072, 3072, 3072, 1024);
    k_gru_gate<<<dim3(2048), dim3(256), 0, stream>>>(gi, gh, hf0, hf0, hb0, 1024);
    // layer 1
    gemm(hb0, 1024, wihb[1], 1024, bih[1], gi, 3072, 3072, 3072, 1024);
    gemm(hb1, 1024, whhb[1], 1024, bhh[1], gh, 3072, 3072, 3072, 1024);
    k_gru_gate<<<dim3(2048), dim3(256), 0, stream>>>(gi, gh, hf1, hf1, hb1, 1024);
    // layer 2 (its bf16 hidden lives in xout[:, :1024], stride 2048)
    gemm(hb1,  1024, wihb[2], 1024, bih[2], gi, 3072, 3072, 3072, 1024);
    gemm(xout, 2048, whhb[2], 1024, bhh[2], gh, 3072, 3072, 3072, 1024);
    k_gru_gate<<<dim3(2048), dim3(256), 0, stream>>>(gi, gh, hf2, hf2, xout, 2048);
    // logits -> d_out[b, t, :100], row stride S*V
    gemm(xout, 2048, outwb, 2048, outbp, out + (size_t)t * V_,
         (long long)S_ * V_, V_, 128, 2048);
  }
}
